// Attention_15745350107179
// MI455X (gfx1250) — matrix operand order repaired and checked by reading, not compiled, not run on hardware
//
#include <hip/hip_runtime.h>
#include <hip/hip_bf16.h>

#define EMBEDD 1024
#define NHEADS 16
#define HD 64

typedef __attribute__((ext_vector_type(16))) _Float16 v16h;
typedef __attribute__((ext_vector_type(8)))  float    v8f;
typedef __attribute__((ext_vector_type(4)))  unsigned v4u;
typedef __attribute__((ext_vector_type(8)))  int      v8i;
typedef __attribute__((ext_vector_type(4)))  int      v4i;

// therock-10.0 headers ship the TDM descriptor header and a 6-arg builtin;
// ROCm 7.2 has the 5-arg form.
#if defined(__has_include)
#if __has_include(<hip/amd_detail/amd_gfx1250_TDM.h>)
#define TDM_6ARG 1
#endif
#endif

union AFrag { unsigned u[8]; v16h v; };

__device__ __forceinline__ v8f wmma_f32_f16(v16h a, v16h b, v8f c) {
  // D = A(16x32 f16) x B(32x16 f16) + C(16x16 f32)
  return __builtin_amdgcn_wmma_f32_16x16x32_f16(
      false, a, false, b, (short)0, c, false, false);
}

#define LDH 36   // padded LDS row stride in halves (16 dwords data + 2 dwords pad)

// ---------------------------------------------------------------------------
// TDM: DMA one [rows=128] x [cols=32] f16 tile (row stride = stride0 elems)
// from global into LDS with hardware row padding of 2 dwords every 16 dwords
// (matches LDH=36).  tensor_dim = remaining extent -> OOB rows read as zero.
// ---------------------------------------------------------------------------
__device__ __forceinline__ void tdm_load_tile_f16(unsigned lds_addr,
                                                  const void* gaddr,
                                                  int dim0_rem, int dim1_rem,
                                                  int stride0) {
  unsigned long long ga = (unsigned long long)gaddr;
  v4u g0;
  g0.x = 1u;                                   // count=1, is_restore=0, gather=0
  g0.y = lds_addr;                             // LDS byte address
  g0.z = (unsigned)ga;                         // global_addr[31:0]
  g0.w = (unsigned)(ga >> 32) | (2u << 30);    // global_addr[56:32] | type=2
  v8i g1;
  g1.s0 = (1 << 16)            // data_size = 2 bytes
        | (1 << 20)            // pad_enable
        | (3 << 22)            // pad_interval: every 16 dwords
        | (1 << 25);           // pad_amount: 2 dwords
  g1.s1 = (dim0_rem & 0xFFFF) << 16;                           // tensor_dim0 lo
  g1.s2 = ((dim0_rem >> 16) & 0xFFFF) | ((dim1_rem & 0xFFFF) << 16);
  g1.s3 = ((dim1_rem >> 16) & 0xFFFF) | (32 << 16);            // tile_dim0 = 32
  g1.s4 = 128;                                                 // tile_dim1 = 128
  g1.s5 = stride0;                                             // dim0 stride lo
  g1.s6 = 0;
  g1.s7 = 0;
  v4i z4 = (v4i)0;
#if defined(TDM_6ARG)
  __builtin_amdgcn_tensor_load_to_lds(g0, g1, z4, z4, (v8i)0, 0);
#else
  __builtin_amdgcn_tensor_load_to_lds(g0, g1, z4, z4, 0);
#endif
}

__device__ __forceinline__ unsigned lds_off(const void* p) {
  return (unsigned)(unsigned long long)p;      // flat shared: low 32 = LDS addr
}

// ---------------------------------------------------------------------------
// Pack: hidden f32->f16, fuse q_w/k_w/v_w -> [3072,1024] f16, out_w f16,
// fused bias [3072] = {q_b, 0, v_b}
// ---------------------------------------------------------------------------
__global__ void pack_kernel(const float* __restrict__ hidden,
                            const float* __restrict__ qw,
                            const float* __restrict__ kw,
                            const float* __restrict__ vw,
                            const float* __restrict__ ow,
                            const float* __restrict__ qb,
                            const float* __restrict__ vb,
                            _Float16* __restrict__ hid_h,
                            _Float16* __restrict__ wqkv_h,
                            _Float16* __restrict__ wout_h,
                            float* __restrict__ bias_qkv,
                            int T) {
  long long i = (long long)blockIdx.x * blockDim.x + threadIdx.x;
  long long n0 = (long long)T * EMBEDD;
  long long n1 = 3LL * EMBEDD * EMBEDD;
  long long n2 = (long long)EMBEDD * EMBEDD;
  long long n3 = 3LL * EMBEDD;
  if (i < n0) {
    hid_h[i] = (_Float16)hidden[i];
  } else if (i < n0 + n1) {
    long long j = i - n0;
    int n = (int)(j >> 10), k = (int)(j & 1023);
    float w;
    if (n < EMBEDD)            w = qw[(size_t)n * EMBEDD + k];
    else if (n < 2 * EMBEDD)   w = kw[(size_t)(n - EMBEDD) * EMBEDD + k];
    else                       w = vw[(size_t)(n - 2 * EMBEDD) * EMBEDD + k];
    wqkv_h[j] = (_Float16)w;
  } else if (i < n0 + n1 + n2) {
    long long j = i - n0 - n1;
    wout_h[j] = (_Float16)ow[j];
  } else if (i < n0 + n1 + n2 + n3) {
    int n = (int)(i - n0 - n1 - n2);
    float bval;
    if (n < EMBEDD)            bval = qb[n];
    else if (n < 2 * EMBEDD)   bval = 0.0f;
    else                       bval = vb[n - 2 * EMBEDD];
    bias_qkv[n] = bval;
  }
}

// ---------------------------------------------------------------------------
// WMMA GEMM with TDM double buffering:  C[M,N] = A[M,K] * W[N,K]^T + bias[N]
// 128x128 tile, 256 threads (8 waves, 2x4), wave tile 64x32, K-step 32.
// Tiles are DMA'd into LDS by the Tensor Data Mover (wave 0 issues, whole
// block consumes after s_wait_tensorcnt + barrier); DMA of step i+1 overlaps
// the 8 WMMAs of step i.
// ---------------------------------------------------------------------------
__global__ __launch_bounds__(256)
void gemm_wmma_kernel(const _Float16* __restrict__ A,
                      const _Float16* __restrict__ W,
                      const float* __restrict__ bias,
                      _Float16* __restrict__ Ch,
                      float* __restrict__ Cf,
                      int M, int N, int K) {
  __shared__ _Float16 lds[2][2][128 * LDH];   // [buffer][A=0/B=1]

  const int tid  = threadIdx.x;
  const int lane = tid & 31;
  const int wid  = tid >> 5;
  const int wm   = (wid >> 2) * 64;   // 0 / 64
  const int wn   = (wid & 3) * 32;    // 0 / 32 / 64 / 96
  const int hk   = lane >> 4;
  const int l15  = lane & 15;
  const int mbase = blockIdx.y * 128;
  const int nbase = blockIdx.x * 128;

  v8f acc[4][2];
#pragma unroll
  for (int mt = 0; mt < 4; ++mt)
#pragma unroll
    for (int nt = 0; nt < 2; ++nt) acc[mt][nt] = (v8f)0.0f;

  const _Float16* Abase = A + (size_t)mbase * K;
  const _Float16* Wbase = W + (size_t)nbase * K;
  const int mrem = M - mbase;          // OOB rows -> TDM returns zeros
  const int nrem = N - nbase;
  const int nkb = K >> 5;

  // prologue: fill buffer 0
  if (wid == 0) {
    tdm_load_tile_f16(lds_off(&lds[0][0][0]), Abase, K, mrem, K);
    tdm_load_tile_f16(lds_off(&lds[0][1][0]), Wbase, K, nrem, K);
    __builtin_amdgcn_s_wait_tensorcnt(0);
  }
  __syncthreads();

  for (int i = 0; i < nkb; ++i) {
    const int cur = i & 1;
    if (wid == 0 && i + 1 < nkb) {     // kick DMA for next K slab
      int kb2 = (i + 1) << 5;
      tdm_load_tile_f16(lds_off(&lds[1 - cur][0][0]), Abase + kb2,
                        K - kb2, mrem, K);
      tdm_load_tile_f16(lds_off(&lds[1 - cur][1][0]), Wbase + kb2,
                        K - kb2, nrem, K);
    }

    const unsigned* lAu = (const unsigned*)&lds[cur][0][0];
    const unsigned* lBu = (const unsigned*)&lds[cur][1][0];

    // B fragments (32x16): lane = N, dword col = v + 8*half (K pairs)
    AFrag bf[2];
#pragma unroll
    for (int nt = 0; nt < 2; ++nt) {
      int bn = wn + nt * 16 + l15;
#pragma unroll
      for (int v = 0; v < 8; ++v)
        bf[nt].u[v] = lBu[bn * (LDH / 2) + v + (hk << 3)];
    }

#pragma unroll
    for (int mt = 0; mt < 4; ++mt) {
      // A fragment (16x32): lane = M, K pairs {0..7}/{8..15}
      AFrag af;
      int am = wm + mt * 16 + l15;
#pragma unroll
      for (int v = 0; v < 8; ++v) {
        int dcol = v + (hk << 3);
        af.u[v] = lAu[am * (LDH / 2) + dcol];
      }
#pragma unroll
      for (int nt = 0; nt < 2; ++nt)
        acc[mt][nt] = wmma_f32_f16(af.v, bf[nt].v, acc[mt][nt]);
    }

    if (wid == 0) __builtin_amdgcn_s_wait_tensorcnt(0);
    __syncthreads();                    // publish next buffer / retire reads
  }

  // ---- epilogue: C layout lane = N, VGPR v = row (v + 8*half) ----
#pragma unroll
  for (int mt = 0; mt < 4; ++mt) {
#pragma unroll
    for (int nt = 0; nt < 2; ++nt) {
#pragma unroll
      for (int v = 0; v < 8; ++v) {
        int gm = mbase + wm + mt * 16 + v + hk * 8;
        int gn = nbase + wn + nt * 16 + l15;
        if (gm < M) {
          float val = acc[mt][nt][v] + bias[gn];
          if (Ch) Ch[(size_t)gm * N + gn] = (_Float16)val;
          else    Cf[(size_t)gm * N + gn] = val;
        }
      }
    }
  }
}

// ---------------------------------------------------------------------------
// In-place RoPE on the f16 QKV buffer [T, 3072] (Q at 0, K at 1024).
// ---------------------------------------------------------------------------
__global__ void rope_kernel(_Float16* __restrict__ qkv,
                            const float* __restrict__ cosb,
                            const float* __restrict__ sinb, int T) {
  int i = blockIdx.x * blockDim.x + threadIdx.x;   // (t, h, d<32)
  if (i >= T * NHEADS * 32) return;
  int d = i & 31;
  int h = (i >> 5) & (NHEADS - 1);
  int t = i >> 9;
  float c = cosb[t * HD + d];
  float s = sinb[t * HD + d];
  size_t base = (size_t)t * 3072 + h * HD + d;
#pragma unroll
  for (int part = 0; part < 2; ++part) {           // Q then K
    float x1 = (float)qkv[base], x2 = (float)qkv[base + 32];
    qkv[base]      = (_Float16)(x1 * c - x2 * s);
    qkv[base + 32] = (_Float16)(x2 * c + x1 * s);
    base += EMBEDD;
  }
}

// ---------------------------------------------------------------------------
// Flash attention: block = (qtile of 64 rows, head, seq), 4 waves x 16 rows.
// qkv: [T,3072] f16 (Q@0 rotated, K@1024 rotated, V@2048). attn: [T,1024] f16.
// ---------------------------------------------------------------------------
__global__ __launch_bounds__(128)
void flash_attn_kernel(const _Float16* __restrict__ qkv,
                       _Float16* __restrict__ attn,
                       const int* __restrict__ seq_len, int B) {
  __shared__ _Float16 lP[4][16 * LDH];

  const int b = blockIdx.z, h = blockIdx.y, qt = blockIdx.x;
  int tok0 = 0;
  for (int i = 0; i < b; ++i) tok0 += seq_len[i];
  const int len = seq_len[b];
  const int q0 = qt * 64;
  if (q0 >= len) return;

  const int wave = threadIdx.x >> 5;
  const int lane = threadIdx.x & 31;
  const int hk   = lane >> 4;
  const int l15  = lane & 15;
  const int qrow = q0 + wave * 16;

  const unsigned* Qu = (const unsigned*)qkv;        // row stride 1536 dwords
  const unsigned short* Vs = (const unsigned short*)qkv;

  AFrag qf[2];
  {
    size_t rowd = ((size_t)(tok0 + qrow + l15) * 3072 + h * HD) >> 1;
#pragma unroll
    for (int s2 = 0; s2 < 2; ++s2)
#pragma unroll
      for (int v = 0; v < 8; ++v) {
        int dcol = v + (hk << 3);
        qf[s2].u[v] = Qu[rowd + s2 * 16 + dcol];
      }
  }

  v8f o[4];
#pragma unroll
  for (int nt = 0; nt < 4; ++nt) o[nt] = (v8f)0.0f;
  float rmax[8], rsum[8];
#pragma unroll
  for (int v = 0; v < 8; ++v) { rmax[v] = -1e30f; rsum[v] = 0.0f; }

  const float scale = 0.125f;                        // 1/sqrt(64)
  const int kend = qrow + 16;                        // causal bound, this wave

  for (int kb = 0; kb < kend; kb += 32) {
    int key0 = kb + l15;       if (key0 >= len) key0 = len - 1;
    int key1 = kb + 16 + l15;  if (key1 >= len) key1 = len - 1;
    size_t r0 = ((size_t)(tok0 + key0) * 3072 + EMBEDD + h * HD) >> 1;
    size_t r1 = ((size_t)(tok0 + key1) * 3072 + EMBEDD + h * HD) >> 1;

    v8f s0 = (v8f)0.0f, s1 = (v8f)0.0f;
#pragma unroll
    for (int s2 = 0; s2 < 2; ++s2) {
      AFrag kf0, kf1;
#pragma unroll
      for (int v = 0; v < 8; ++v) {
        kf0.u[v] = Qu[r0 + s2 * 16 + (hk << 3) + v];
        kf1.u[v] = Qu[r1 + s2 * 16 + (hk << 3) + v];
      }
      s0 = wmma_f32_f16(qf[s2].v, kf0.v, s0);
      s1 = wmma_f32_f16(qf[s2].v, kf1.v, s1);
    }

    float p0[8], p1[8];
#pragma unroll
    for (int v = 0; v < 8; ++v) {
      int qg = qrow + v + hk * 8;
      float x0 = s0[v] * scale, x1 = s1[v] * scale;
      if (kb + l15 > qg)      x0 = -1e30f;
      if (kb + 16 + l15 > qg) x1 = -1e30f;
      float mv = fmaxf(x0, x1);
#pragma unroll
      for (int off = 1; off < 16; off <<= 1)
        mv = fmaxf(mv, __shfl_xor(mv, off, 32));
      float mN = fmaxf(rmax[v], mv);
      float alpha = __expf(rmax[v] - mN);
      rmax[v] = mN;
      float e0 = __expf(x0 - mN), e1 = __expf(x1 - mN);
      float rs = e0 + e1;
#pragma unroll
      for (int off = 1; off < 16; off <<= 1) rs += __shfl_xor(rs, off, 32);
      rsum[v] = rsum[v] * alpha + rs;
#pragma unroll
      for (int nt = 0; nt < 4; ++nt) o[nt][v] *= alpha;
      p0[v] = e0; p1[v] = e1;
    }

    // transpose P (C layout) -> A layout via per-wave LDS
    _Float16* pw = lP[wave];
#pragma unroll
    for (int v = 0; v < 8; ++v) {
      int m = v + hk * 8;
      pw[m * LDH + l15]      = (_Float16)p0[v];
      pw[m * LDH + 16 + l15] = (_Float16)p1[v];
    }
    AFrag pf;
    const unsigned* pu = (const unsigned*)pw;
#pragma unroll
    for (int v = 0; v < 8; ++v) {
      int dcol = v + (hk << 3);
      pf.u[v] = pu[l15 * (LDH / 2) + dcol];
    }

    // O += P * V   (B frag: lane = dh, VGPR v = key pair kb + 16*hk + 2v)
#pragma unroll
    for (int nt = 0; nt < 4; ++nt) {
      AFrag vf;
      int dh = nt * 16 + l15;
#pragma unroll
      for (int v = 0; v < 8; ++v) {
        int key = kb + (hk << 4) + 2 * v;
        int k0 = key < len ? key : len - 1;
        int k1 = key + 1 < len ? key + 1 : len - 1;
        unsigned lo = Vs[(size_t)(tok0 + k0) * 3072 + 2 * EMBEDD + h * HD + dh];
        unsigned hi = Vs[(size_t)(tok0 + k1) * 3072 + 2 * EMBEDD + h * HD + dh];
        vf.u[v] = lo | (hi << 16);
      }
      o[nt] = wmma_f32_f16(pf.v, vf.v, o[nt]);
    }
  }

#pragma unroll
  for (int v = 0; v < 8; ++v) {
    float inv = 1.0f / rsum[v];
    int m = v + hk * 8;
    size_t rowo = (size_t)(tok0 + qrow + m) * EMBEDD + h * HD;
#pragma unroll
    for (int nt = 0; nt < 4; ++nt)
      attn[rowo + nt * 16 + l15] = (_Float16)(o[nt][v] * inv);
  }
}

// ---------------------------------------------------------------------------
extern "C" void kernel_launch(void* const* d_in, const int* in_sizes, int n_in,
                              void* d_out, int out_size, void* d_ws, size_t ws_size,
                              hipStream_t stream) {
  const float* hidden = (const float*)d_in[0];
  const float* cosb   = (const float*)d_in[1];
  const float* sinb   = (const float*)d_in[2];
  const float* qw     = (const float*)d_in[3];
  const float* qb     = (const float*)d_in[4];
  const float* kw     = (const float*)d_in[5];
  const float* vw     = (const float*)d_in[6];
  const float* vb     = (const float*)d_in[7];
  const float* ow     = (const float*)d_in[8];
  const float* ob     = (const float*)d_in[9];
  const int* seq_len  = (const int*)d_in[10];
  const int T = in_sizes[0] / EMBEDD;     // 9792 (multiple of 64)
  const int B = in_sizes[10];             // 12
  float* out = (float*)d_out;

  char* p = (char*)d_ws;
  _Float16* hid_h  = (_Float16*)p; p += (size_t)T * EMBEDD * 2;
  _Float16* wqkv_h = (_Float16*)p; p += (size_t)3 * EMBEDD * EMBEDD * 2;
  _Float16* wout_h = (_Float16*)p; p += (size_t)EMBEDD * EMBEDD * 2;
  float*    bias_q = (float*)p;    p += (size_t)3 * EMBEDD * 4;
  _Float16* qkv_h  = (_Float16*)p; p += (size_t)T * 3 * EMBEDD * 2;
  _Float16* attn_h = (_Float16*)p; p += (size_t)T * EMBEDD * 2;

  // 1) pack/convert
  long long npack = (long long)T * EMBEDD + 4LL * EMBEDD * EMBEDD + 3LL * EMBEDD;
  pack_kernel<<<(unsigned)((npack + 255) / 256), 256, 0, stream>>>(
      hidden, qw, kw, vw, ow, qb, vb, hid_h, wqkv_h, wout_h, bias_q, T);

  // 2) fused QKV projection: [T,1024] x [3072,1024]^T + bias -> f16 [T,3072]
  dim3 g1(3 * EMBEDD / 128, (T + 127) / 128);
  gemm_wmma_kernel<<<g1, 256, 0, stream>>>(hid_h, wqkv_h, bias_q, qkv_h, nullptr,
                                           T, 3 * EMBEDD, EMBEDD);

  // 3) RoPE in-place on Q,K
  rope_kernel<<<(unsigned)(((long long)T * NHEADS * 32 + 255) / 256), 256, 0,
                stream>>>(qkv_h, cosb, sinb, T);

  // 4) causal varlen flash attention
  dim3 g2(16 /* 1024/64 qtiles */, NHEADS, B);
  flash_attn_kernel<<<g2, 128, 0, stream>>>(qkv_h, attn_h, seq_len, B);

  // 5) output projection -> f32 d_out
  dim3 g3(EMBEDD / 128, (T + 127) / 128);
  gemm_wmma_kernel<<<g3, 256, 0, stream>>>(attn_h, wout_h, ob, nullptr, out,
                                           T, EMBEDD, EMBEDD);
}
